// _SinkhornDistance_81011673137751
// MI455X (gfx1250) — matrix operand order repaired and checked by reading, not compiled, not run on hardware
//
#include <hip/hip_runtime.h>
#include <hip/hip_bf16.h>
#include <stdint.h>

typedef __attribute__((ext_vector_type(16))) _Float16 v16h;
typedef __attribute__((ext_vector_type(4)))  _Float16 v4h;
typedef __attribute__((ext_vector_type(8)))  float    v8f;

#define NPT      8192
#define DIM      64
#define NT       (NPT / 16)          /* 512 column tiles */
#define EPSV     0.05f
#define INV_EPS  20.0f
#define LOG_W    (-9.0109131782f)    /* log(1/8192); 1/8192 > 1e-6 so no clamp */
#define THRESH   1e-3f
#define MAX_ITER 50

#define USE_ASYNC_COPY 1

// ---------------------------------------------------------------------------
// Prep: split f32 -> (hi,lo) f16 pair so 3 WMMAs emulate an f32 GEMM.
// One float4 per thread.
// ---------------------------------------------------------------------------
__global__ __launch_bounds__(256) void prep_convert(
    const float* __restrict__ S, const float* __restrict__ T,
    _Float16* __restrict__ shi, _Float16* __restrict__ slo,
    _Float16* __restrict__ thi, _Float16* __restrict__ tlo) {
  int i = blockIdx.x * blockDim.x + threadIdx.x;      // 0 .. 2*NPT*DIM/4-1
  const float4* src;
  _Float16 *dhi, *dlo;
  int j;
  if (i < NPT * DIM / 4) { src = (const float4*)S; j = i;                 dhi = shi; dlo = slo; }
  else                   { src = (const float4*)T; j = i - NPT * DIM / 4; dhi = thi; dlo = tlo; }
  float4 x = src[j];
  v4h h, l;
  h[0] = (_Float16)x.x; l[0] = (_Float16)(x.x - (float)h[0]);
  h[1] = (_Float16)x.y; l[1] = (_Float16)(x.y - (float)h[1]);
  h[2] = (_Float16)x.z; l[2] = (_Float16)(x.z - (float)h[2]);
  h[3] = (_Float16)x.w; l[3] = (_Float16)(x.w - (float)h[3]);
  *(v4h*)(dhi + j * 4) = h;
  *(v4h*)(dlo + j * 4) = l;
}

// Row squared-norms (16 lanes per row, float4 each) + zero state.
__global__ __launch_bounds__(256) void prep_sq(
    const float* __restrict__ S, const float* __restrict__ T,
    float* __restrict__ sqs, float* __restrict__ sqt,
    float* __restrict__ u, float* __restrict__ v,
    float* __restrict__ ad, int* flag, float* out) {
  int g   = (blockIdx.x * blockDim.x + threadIdx.x) >> 4;  // row 0 .. 2*NPT-1
  int sub = threadIdx.x & 15;
  const float* base = (g < NPT) ? (S + g * DIM) : (T + (g - NPT) * DIM);
  float4 x = ((const float4*)base)[sub];
  float s = x.x * x.x + x.y * x.y + x.z * x.z + x.w * x.w;
  #pragma unroll
  for (int off = 1; off < 16; off <<= 1) s += __shfl_xor(s, off, 32);
  if (sub == 0) {
    if (g < NPT) { sqs[g] = s; u[g] = 0.f; ad[g] = 0.f; }
    else         { sqt[g - NPT] = s; v[g - NPT] = 0.f; }
    if (g == 0) { *flag = 0; *out = 0.f; }
  }
}

// ---------------------------------------------------------------------------
// Fragment loaders matching the CDNA5 WMMA 16x16x32 f16 VGPR layouts
// (CDNA5 ISA §7.12.2).
// A (16x32, MxK): lane half h, row m: V0..3 hold K = kt*32 + h*8 + {0..7},
//                 V4..7 hold K = kt*32 + 16 + h*8 + {0..7}  -> two 8-elem runs.
// B (32x16, KxN): lane half h, col m: V0..3 hold K = kt*32 + h*8 + {0..7},
//                 V4..7 hold K = kt*32 + 16 + h*8 + {0..7}  -> two 8-elem runs.
// ---------------------------------------------------------------------------
__device__ __forceinline__ void load_a_frags(const _Float16* __restrict__ Ahi,
                                             const _Float16* __restrict__ Alo,
                                             int arow, int hf, v16h ahi[2], v16h alo[2]) {
  #pragma unroll
  for (int kt = 0; kt < 2; ++kt) {
    const int kb = kt * 32 + hf * 8;
    const _Float16* ph = Ahi + arow * DIM + kb;
    const _Float16* pl = Alo + arow * DIM + kb;
    #pragma unroll
    for (int e = 0; e < 8; ++e) {
      ahi[kt][e]     = ph[e];
      ahi[kt][8 + e] = ph[16 + e];
      alo[kt][e]     = pl[e];
      alo[kt][8 + e] = pl[16 + e];
    }
  }
}

__device__ __forceinline__ void load_b_frags(const _Float16* lhi, const _Float16* llo,
                                             int m, int hf, v16h bhi[2], v16h blo[2]) {
  #pragma unroll
  for (int kt = 0; kt < 2; ++kt) {
    const int kb = kt * 32 + hf * 8;
    const _Float16* ph = lhi + m * DIM + kb;
    const _Float16* pl = llo + m * DIM + kb;
    #pragma unroll
    for (int e = 0; e < 8; ++e) {
      bhi[kt][e] = ph[e]; bhi[kt][8 + e] = ph[16 + e];
      blo[kt][e] = pl[e]; blo[kt][8 + e] = pl[16 + e];
    }
  }
}

// Split-f16 emulated f32 GEMM over K=64; two accumulator chains so the
// WMMA RAW dependency depth is halved (hi*hi chain || cross-term chain).
__device__ __forceinline__ void dot_wmma(const v16h ahi[2], const v16h alo[2],
                                         const v16h bhi[2], const v16h blo[2],
                                         v8f& acc1, v8f& acc2) {
  acc1 = (v8f){};
  acc2 = (v8f){};
  #pragma unroll
  for (int kt = 0; kt < 2; ++kt) {
    acc1 = __builtin_amdgcn_wmma_f32_16x16x32_f16(false, ahi[kt], false, bhi[kt],
                                                  (short)0, acc1, false, false);
    acc2 = __builtin_amdgcn_wmma_f32_16x16x32_f16(false, ahi[kt], false, blo[kt],
                                                  (short)0, acc2, false, false);
    acc2 = __builtin_amdgcn_wmma_f32_16x16x32_f16(false, alo[kt], false, bhi[kt],
                                                  (short)0, acc2, false, false);
  }
}

// Issue async global->LDS copy of one 16-row tile (hi+lo, 2KB each; 256 thr).
__device__ __forceinline__ void stage_issue(const _Float16* __restrict__ Bhi,
                                            const _Float16* __restrict__ Blo,
                                            _Float16* lhi, _Float16* llo,
                                            int c, int tid) {
  const _Float16* gh = Bhi + c * 16 * DIM + tid * 4;   // 4 halfwords / thread
  const _Float16* gl = Blo + c * 16 * DIM + tid * 4;
#if USE_ASYNC_COPY
  unsigned dh = (unsigned)(uintptr_t)&lhi[tid * 4];
  unsigned dl = (unsigned)(uintptr_t)&llo[tid * 4];
  unsigned long long ah = (unsigned long long)(uintptr_t)gh;
  unsigned long long al = (unsigned long long)(uintptr_t)gl;
  asm volatile("global_load_async_to_lds_b64 %0, %2, off\n\t"
               "global_load_async_to_lds_b64 %1, %3, off"
               :: "v"(dh), "v"(dl), "v"(ah), "v"(al)
               : "memory");
#else
  *(uint2*)&lhi[tid * 4] = *(const uint2*)gh;
  *(uint2*)&llo[tid * 4] = *(const uint2*)gl;
#endif
}

#if USE_ASYNC_COPY
#define ASYNC_WAIT() asm volatile("s_wait_asynccnt 0" ::: "memory")
#else
#define ASYNC_WAIT() ((void)0)
#endif

// ---------------------------------------------------------------------------
// One Sinkhorn half-step:
//   p[x] <- EPS*(LOG_W - LSE_y((p[x]+q[y]-C(x,y))/EPS)) + p[x]
// A side = "x" rows (16 per wave), B side = "y" tiles double-buffered through
// LDS via async DMA. Flash-style online logsumexp, 16-lane shuffle merge.
// ---------------------------------------------------------------------------
__global__ __launch_bounds__(256) void lse_pass(
    const _Float16* __restrict__ Ahi, const _Float16* __restrict__ Alo,
    const _Float16* __restrict__ Bhi, const _Float16* __restrict__ Blo,
    const float* __restrict__ sqA, const float* __restrict__ sqB,
    const float* __restrict__ q, float* __restrict__ p,
    float* __restrict__ absdiff, const int* __restrict__ flag) {
  if (*flag) return;  // frozen after convergence (uniform -> barrier-safe)

  __shared__ __align__(16) _Float16 lhi[2][16 * DIM];
  __shared__ __align__(16) _Float16 llo[2][16 * DIM];

  const int tid  = threadIdx.x;
  const int wave = tid >> 5;
  const int lane = tid & 31;
  const int hf   = lane >> 4;
  const int m    = lane & 15;
  const int r0   = blockIdx.x * 128 + wave * 16;

  v16h ahi[2], alo[2];
  load_a_frags(Ahi, Alo, r0 + m, hf, ahi, alo);

  float pr[8], sa[8];
  #pragma unroll
  for (int e = 0; e < 8; ++e) {
    int rr = r0 + e + 8 * hf;   // C/D row for this lane/VGPR slot
    pr[e] = p[rr];
    sa[e] = sqA[rr];
  }

  float mx[8], sm[8];
  #pragma unroll
  for (int e = 0; e < 8; ++e) { mx[e] = -3.0e38f; sm[e] = 0.0f; }

  stage_issue(Bhi, Blo, lhi[0], llo[0], 0, tid);   // prologue: tile 0 in flight
  float qc = q[m];                                  // tile-0 scalars (col = m)
  float sb = sqB[m];

  for (int c = 0; c < NT; ++c) {
    const int cur = c & 1;
    ASYNC_WAIT();          // own slice of tile c landed
    __syncthreads();       // whole tile c in LDS; buf[1-cur] free (readers done)
    if (c + 1 < NT) stage_issue(Bhi, Blo, lhi[1 - cur], llo[1 - cur], c + 1, tid);

    float qn = 0.f, sn = 0.f;                       // prefetch next-tile scalars
    if (c + 1 < NT) { int nc = (c + 1) * 16 + m; qn = q[nc]; sn = sqB[nc]; }

    v16h bhi[2], blo[2];
    load_b_frags(lhi[cur], llo[cur], m, hf, bhi, blo);
    v8f acc1, acc2;
    dot_wmma(ahi, alo, bhi, blo, acc1, acc2);

    #pragma unroll
    for (int e = 0; e < 8; ++e) {
      float dot  = acc1[e] + acc2[e];
      float cost = fmaxf(sa[e] + sb - 2.0f * dot, 0.0f);
      float x  = (pr[e] + qc - cost) * INV_EPS;
      float nm = fmaxf(mx[e], x);
      sm[e] = sm[e] * __expf(mx[e] - nm) + __expf(x - nm);
      mx[e] = nm;
    }
    qc = qn; sb = sn;
  }

  // merge the 16 per-lane partial LSEs within each half-wave
  #pragma unroll
  for (int off = 1; off < 16; off <<= 1) {
    #pragma unroll
    for (int e = 0; e < 8; ++e) {
      float om = __shfl_xor(mx[e], off, 32);
      float os = __shfl_xor(sm[e], off, 32);
      float nm = fmaxf(mx[e], om);
      sm[e] = sm[e] * __expf(mx[e] - nm) + os * __expf(om - nm);
      mx[e] = nm;
    }
  }

  if (m == 0) {
    #pragma unroll
    for (int e = 0; e < 8; ++e) {
      int rr = r0 + e + 8 * hf;
      float lse = mx[e] + __logf(sm[e]);
      float np  = EPSV * (LOG_W - lse) + pr[e];
      p[rr] = np;
      if (absdiff) absdiff[rr] = fabsf(np - pr[e]);
    }
  }
}

// mean(|u_new - u_prev|) < THRESH  -> freeze (emulates the torch early break)
__global__ __launch_bounds__(256) void conv_check(const float* __restrict__ ad, int* flag) {
  __shared__ float red[256];
  float s = 0.f;
  for (int i = threadIdx.x; i < NPT; i += 256) s += ad[i];
  red[threadIdx.x] = s;
  __syncthreads();
  for (int o = 128; o > 0; o >>= 1) {
    if (threadIdx.x < o) red[threadIdx.x] += red[threadIdx.x + o];
    __syncthreads();
  }
  if (threadIdx.x == 0 && (red[0] * (1.0f / NPT)) < THRESH) *flag = 1;
}

// Per-wave partials of sum( exp((-C+u+v)/eps) * C ) -> part[512] (deterministic)
__global__ __launch_bounds__(256) void final_sum(
    const _Float16* __restrict__ Ahi, const _Float16* __restrict__ Alo,
    const _Float16* __restrict__ Bhi, const _Float16* __restrict__ Blo,
    const float* __restrict__ sqA, const float* __restrict__ sqB,
    const float* __restrict__ u, const float* __restrict__ v,
    float* __restrict__ part_out) {
  __shared__ __align__(16) _Float16 lhi[2][16 * DIM];
  __shared__ __align__(16) _Float16 llo[2][16 * DIM];

  const int tid  = threadIdx.x;
  const int wave = tid >> 5;
  const int lane = tid & 31;
  const int hf   = lane >> 4;
  const int m    = lane & 15;
  const int r0   = blockIdx.x * 128 + wave * 16;

  v16h ahi[2], alo[2];
  load_a_frags(Ahi, Alo, r0 + m, hf, ahi, alo);

  float ur[8], sa[8];
  #pragma unroll
  for (int e = 0; e < 8; ++e) {
    int rr = r0 + e + 8 * hf;
    ur[e] = u[rr];
    sa[e] = sqA[rr];
  }

  stage_issue(Bhi, Blo, lhi[0], llo[0], 0, tid);
  float vc = v[m];
  float sb = sqB[m];

  float part = 0.f;
  for (int c = 0; c < NT; ++c) {
    const int cur = c & 1;
    ASYNC_WAIT();
    __syncthreads();
    if (c + 1 < NT) stage_issue(Bhi, Blo, lhi[1 - cur], llo[1 - cur], c + 1, tid);

    float vn = 0.f, sn = 0.f;
    if (c + 1 < NT) { int nc = (c + 1) * 16 + m; vn = v[nc]; sn = sqB[nc]; }

    v16h bhi[2], blo[2];
    load_b_frags(lhi[cur], llo[cur], m, hf, bhi, blo);
    v8f acc1, acc2;
    dot_wmma(ahi, alo, bhi, blo, acc1, acc2);

    #pragma unroll
    for (int e = 0; e < 8; ++e) {
      float dot  = acc1[e] + acc2[e];
      float cost = fmaxf(sa[e] + sb - 2.0f * dot, 0.0f);
      float plan = __expf((ur[e] + vc - cost) * INV_EPS);
      part += plan * cost;
    }
    vc = vn; sb = sn;
  }

  #pragma unroll
  for (int off = 1; off < 32; off <<= 1) part += __shfl_xor(part, off, 32);
  if (lane == 0) part_out[blockIdx.x * 8 + wave] = part;
}

// Fixed-order tree reduction of the 512 wave partials -> scalar output.
__global__ __launch_bounds__(256) void final_reduce(const float* __restrict__ part,
                                                    float* __restrict__ out) {
  __shared__ float red[256];
  red[threadIdx.x] = part[threadIdx.x] + part[threadIdx.x + 256];
  __syncthreads();
  for (int o = 128; o > 0; o >>= 1) {
    if (threadIdx.x < o) red[threadIdx.x] += red[threadIdx.x + o];
    __syncthreads();
  }
  if (threadIdx.x == 0) *out = red[0];
}

// ---------------------------------------------------------------------------
extern "C" void kernel_launch(void* const* d_in, const int* in_sizes, int n_in,
                              void* d_out, int out_size, void* d_ws, size_t ws_size,
                              hipStream_t stream) {
  const float* S = (const float*)d_in[0];
  const float* T = (const float*)d_in[1];

  char* ws = (char*)d_ws;
  float* u    = (float*)ws;
  float* v    = u + NPT;
  float* ad   = v + NPT;          // |du| during iters; wave partials at the end
  float* sqs  = ad + NPT;
  float* sqt  = sqs + NPT;
  int*   flag = (int*)(sqt + NPT);
  _Float16* shi = (_Float16*)(flag + 16);      // keep 64B alignment
  _Float16* slo = shi + NPT * DIM;
  _Float16* thi = slo + NPT * DIM;
  _Float16* tlo = thi + NPT * DIM;
  float* out = (float*)d_out;

  prep_convert<<<(2 * NPT * DIM / 4) / 256, 256, 0, stream>>>(S, T, shi, slo, thi, tlo);
  prep_sq<<<(2 * NPT * 16) / 256, 256, 0, stream>>>(S, T, sqs, sqt, u, v, ad, flag, out);

  for (int it = 0; it < MAX_ITER; ++it) {
    // u-update: p=u (rows = source), q=v (cols = target), record |du|
    lse_pass<<<NPT / 128, 256, 0, stream>>>(shi, slo, thi, tlo, sqs, sqt, v, u, ad, flag);
    // v-update with fresh u: p=v (rows = target), q=u (cols = source)
    lse_pass<<<NPT / 128, 256, 0, stream>>>(thi, tlo, shi, slo, sqt, sqs, u, v, nullptr, flag);
    conv_check<<<1, 256, 0, stream>>>(ad, flag);
  }

  final_sum<<<NPT / 128, 256, 0, stream>>>(shi, slo, thi, tlo, sqs, sqt, u, v, ad);
  final_reduce<<<1, 256, 0, stream>>>(ad, out);
}
